// AutoRegressiveLSTM_31817117729192
// MI455X (gfx1250) — matrix operand order repaired and checked by reading, not compiled, not run on hardware
//
#include <hip/hip_runtime.h>

typedef __bf16 bf16;
typedef __attribute__((ext_vector_type(8)))  bf16  v8bf;
typedef __attribute__((ext_vector_type(16))) bf16  v16bf;
typedef __attribute__((ext_vector_type(8)))  float v8f;

__device__ __forceinline__ float sigmoidf_(float x) {
    return 1.0f / (1.0f + __expf(-x));
}

// ---------------- weight prep (bf16, transposed to [N][K]) ----------------

// Wc1t[n][k] , n in [0,4096), k in [0,1152):  k<128 -> W1[k][n], else U1[k-128][n]
__global__ __launch_bounds__(256) void pack_w1(const float* __restrict__ W1,
                                               const float* __restrict__ U1,
                                               bf16* __restrict__ out) {
    const int total = 4096 * 1152;
    for (int id = blockIdx.x * 256 + threadIdx.x; id < total; id += gridDim.x * 256) {
        int n = id / 1152, k = id - n * 1152;
        float v = (k < 128) ? W1[k * 4096 + n] : U1[(k - 128) * 4096 + n];
        out[id] = (bf16)v;
    }
}

// cell2 is called with x == h, so z = h@(W2+U2): fold the sum once.
__global__ __launch_bounds__(256) void pack_w2(const float* __restrict__ W2,
                                               const float* __restrict__ U2,
                                               bf16* __restrict__ out) {
    const int total = 4096 * 1024;
    for (int id = blockIdx.x * 256 + threadIdx.x; id < total; id += gridDim.x * 256) {
        int n = id / 1024, k = id - n * 1024;
        out[id] = (bf16)(W2[k * 4096 + n] + U2[k * 4096 + n]);
    }
}

__global__ __launch_bounds__(256) void pack_wd(const float* __restrict__ Wd,
                                               bf16* __restrict__ out) {
    const int total = 128 * 1024;
    for (int id = blockIdx.x * 256 + threadIdx.x; id < total; id += gridDim.x * 256) {
        int n = id / 1024, k = id - n * 1024;
        out[id] = (bf16)Wd[k * 128 + n];
    }
}

// x = ((in - mean)*inv - mean)*inv  (double-normalize, faithful to reference)
__global__ __launch_bounds__(256) void normalize_in(const float* __restrict__ in,
                                                    const float* __restrict__ mean,
                                                    const float* __restrict__ var,
                                                    bf16* __restrict__ out) {
    const int total = 512 * 64 * 128;
    for (int id = blockIdx.x * 256 + threadIdx.x; id < total; id += gridDim.x * 256) {
        int f = id & 127;
        float inv = rsqrtf(var[f] + 1e-7f);
        float m = mean[f];
        float v = (in[id] - m) * inv;
        v = (v - m) * inv;
        out[id] = (bf16)v;
    }
}

__global__ __launch_bounds__(256) void init_state(bf16* __restrict__ h,
                                                  float* __restrict__ c) {
    const int total = 512 * 1024;
    for (int id = blockIdx.x * 256 + threadIdx.x; id < total; id += gridDim.x * 256) {
        h[id] = (bf16)0.0f;
        c[id] = 0.0f;
    }
}

// ---------------- fused LSTM step ----------------
// z = concat(A0,A1) @ W + b, gates i,f,g,o; c' = f*c + i*g; h' = o*tanh(c').
// Wt is bf16, transposed: row (gate*1024 + unit), ldw columns of K.
// Block: 256 thr (8 waves). Tile: 64 rows x 32 units; wave = one (m,n) 16x16
// position with 4 gate accumulators. Grid: (512/64, 1024/32) = (8, 32).
__global__ __launch_bounds__(256) void lstm_step(
    const bf16* __restrict__ A0, int lda0, int K0,
    const bf16* __restrict__ A1, int lda1,
    const bf16* __restrict__ Wt, int ldw, int Ktot,
    const float* __restrict__ bias,
    float* __restrict__ c,
    bf16* __restrict__ h_out)
{
    __shared__ alignas(16) bf16 ldsA[64 * 40];    // 64 rows x 32 K (stride 40)
    __shared__ alignas(16) bf16 ldsB[128 * 40];   // 4 gates x 32 units x 32 K

    const int t    = threadIdx.x;
    const int lane = t & 31;
    const int wv   = t >> 5;
    const int mt   = wv & 3;        // 4 m-subtiles of 16 rows
    const int nt   = wv >> 2;       // 2 n-subtiles of 16 units
    const int kg   = lane >> 4;     // lane half (K-group for A/B frags)
    const int nl   = lane & 15;
    const int m0   = blockIdx.x * 64;
    const int n0   = blockIdx.y * 32;

    v8f acc0 = {0.f,0.f,0.f,0.f,0.f,0.f,0.f,0.f};
    v8f acc1 = acc0, acc2 = acc0, acc3 = acc0;

    // staging assignments
    const int arow = t >> 2;              // 0..63
    const int acq  = t & 3;               // 0..3 -> 8 bf16 each
    const int brow = t >> 1;              // 0..127
    const int sgate = brow >> 5;          // 0..3
    const int bj   = brow & 31;           // unit within strip
    const int bh   = t & 1;               // K-half of 32-chunk
    const long wro = (long)(sgate * 1024 + n0 + bj) * ldw;

    const int nkc = Ktot >> 5;
    for (int kc = 0; kc < nkc; ++kc) {
        const int kbase = kc << 5;
        // stage A chunk (K0 is a multiple of 32 -> chunk never straddles A0/A1)
        {
            int kglob = kbase + 8 * acq;
            const bf16* src = (kglob < K0)
                ? (A0 + (long)(m0 + arow) * lda0 + kglob)
                : (A1 + (long)(m0 + arow) * lda1 + (kglob - K0));
            *(v8bf*)&ldsA[arow * 40 + 8 * acq] = *(const v8bf*)src;
        }
        // stage 128 weight rows (4 gates x 32 units), 32 K each
        {
            const bf16* src = Wt + wro + kbase + 16 * bh;
            *(v8bf*)&ldsB[brow * 40 + 16 * bh]     = *(const v8bf*)(src);
            *(v8bf*)&ldsB[brow * 40 + 16 * bh + 8] = *(const v8bf*)(src + 8);
        }
        __syncthreads();

        // A fragment: lane half kg holds
        // K = {16kg..16kg+15} contiguous in elems 0..15
        const bf16* ap = &ldsA[(16 * mt + nl) * 40];
        v8bf alo = *(const v8bf*)(ap + 16 * kg);
        v8bf ahi = *(const v8bf*)(ap + 16 * kg + 8);
        v16bf af;
        #pragma unroll
        for (int e = 0; e < 8; ++e) { af[e] = alo[e]; af[e + 8] = ahi[e]; }

        // B fragments: lane half kg holds K = 16kg..16kg+15 contiguous
        v16bf bf0 = *(const v16bf*)&ldsB[(0 * 32 + nt * 16 + nl) * 40 + 16 * kg];
        v16bf bf1 = *(const v16bf*)&ldsB[(1 * 32 + nt * 16 + nl) * 40 + 16 * kg];
        v16bf bf2 = *(const v16bf*)&ldsB[(2 * 32 + nt * 16 + nl) * 40 + 16 * kg];
        v16bf bf3 = *(const v16bf*)&ldsB[(3 * 32 + nt * 16 + nl) * 40 + 16 * kg];

        acc0 = __builtin_amdgcn_wmma_f32_16x16x32_bf16(false, af, false, bf0, (short)0, acc0, false, false);
        acc1 = __builtin_amdgcn_wmma_f32_16x16x32_bf16(false, af, false, bf1, (short)0, acc1, false, false);
        acc2 = __builtin_amdgcn_wmma_f32_16x16x32_bf16(false, af, false, bf2, (short)0, acc2, false, false);
        acc3 = __builtin_amdgcn_wmma_f32_16x16x32_bf16(false, af, false, bf3, (short)0, acc3, false, false);
        __syncthreads();
    }

    // fused gate math + state update; C/D layout: VGPR v -> row v + 8*kg
    const int u = n0 + nt * 16 + nl;
    const float bi = bias[u];
    const float bfv = bias[1024 + u];
    const float bgv = bias[2048 + u];
    const float bo = bias[3072 + u];
    const int mBase = m0 + 16 * mt + 8 * kg;
    #pragma unroll
    for (int v = 0; v < 8; ++v) {
        const int idx = (mBase + v) * 1024 + u;
        float ig = sigmoidf_(acc0[v] + bi);
        float fg = sigmoidf_(acc1[v] + bfv);
        float gg = tanhf(acc2[v] + bgv);
        float og = sigmoidf_(acc3[v] + bo);
        float cn = fg * c[idx] + ig * gg;
        c[idx] = cn;
        h_out[idx] = (bf16)(og * tanhf(cn));
    }
}

// ---------------- dense projection: pred = h @ Wd + bd ----------------
// Writes f32 into out[b][s][f] and bf16 into predb (next AR input).
// Grid (8, 4); block 256 = 8 waves; tile 64 x 32.
__global__ __launch_bounds__(256) void dense_step(
    const bf16* __restrict__ h, const bf16* __restrict__ Wdt,
    const float* __restrict__ bd,
    float* __restrict__ out, bf16* __restrict__ predb, int s)
{
    __shared__ alignas(16) bf16 ldsA[64 * 40];
    __shared__ alignas(16) bf16 ldsB[32 * 40];

    const int t = threadIdx.x;
    const int lane = t & 31, wv = t >> 5;
    const int mt = wv & 3, nt = wv >> 2;
    const int kg = lane >> 4, nl = lane & 15;
    const int m0 = blockIdx.x * 64, n0 = blockIdx.y * 32;

    v8f acc = {0.f,0.f,0.f,0.f,0.f,0.f,0.f,0.f};
    const int arow = t >> 2, acq = t & 3;

    for (int kc = 0; kc < 32; ++kc) {
        const int kbase = kc << 5;
        *(v8bf*)&ldsA[arow * 40 + 8 * acq] =
            *(const v8bf*)(h + (long)(m0 + arow) * 1024 + kbase + 8 * acq);
        if (t < 128) {
            int r = t >> 2, cq = t & 3;
            *(v8bf*)&ldsB[r * 40 + 8 * cq] =
                *(const v8bf*)(Wdt + (long)(n0 + r) * 1024 + kbase + 8 * cq);
        }
        __syncthreads();

        const bf16* ap = &ldsA[(16 * mt + nl) * 40];
        v8bf alo = *(const v8bf*)(ap + 16 * kg);
        v8bf ahi = *(const v8bf*)(ap + 16 * kg + 8);
        v16bf af;
        #pragma unroll
        for (int e = 0; e < 8; ++e) { af[e] = alo[e]; af[e + 8] = ahi[e]; }

        v16bf bf_ = *(const v16bf*)&ldsB[(nt * 16 + nl) * 40 + 16 * kg];
        acc = __builtin_amdgcn_wmma_f32_16x16x32_bf16(false, af, false, bf_, (short)0, acc, false, false);
        __syncthreads();
    }

    const int n = n0 + nt * 16 + nl;
    const float bv = bd[n];
    const int mBase = m0 + 16 * mt + 8 * kg;
    #pragma unroll
    for (int v = 0; v < 8; ++v) {
        const int m = mBase + v;
        float val = acc[v] + bv;
        out[((long)m * 32 + s) * 128 + n] = val;
        predb[m * 128 + n] = (bf16)val;
    }
}

// ---------------- host orchestration ----------------
extern "C" void kernel_launch(void* const* d_in, const int* in_sizes, int n_in,
                              void* d_out, int out_size, void* d_ws, size_t ws_size,
                              hipStream_t stream) {
    (void)in_sizes; (void)n_in; (void)out_size; (void)ws_size;
    const float* inputs = (const float*)d_in[0];
    const float* mean   = (const float*)d_in[1];
    const float* var    = (const float*)d_in[2];
    const float* W1 = (const float*)d_in[3];
    const float* U1 = (const float*)d_in[4];
    const float* b1 = (const float*)d_in[5];
    const float* W2 = (const float*)d_in[6];
    const float* U2 = (const float*)d_in[7];
    const float* b2 = (const float*)d_in[8];
    const float* Wd = (const float*)d_in[9];
    const float* bd = (const float*)d_in[10];
    float* out = (float*)d_out;

    char* ws = (char*)d_ws;
    bf16* Wc1t  = (bf16*)ws;  ws += (size_t)4096 * 1152 * 2;  // [4N][1152]
    bf16* Ws2t  = (bf16*)ws;  ws += (size_t)4096 * 1024 * 2;  // [4N][1024] (W2+U2)^T
    bf16* Wdt   = (bf16*)ws;  ws += (size_t)128  * 1024 * 2;  // [128][1024]
    bf16* xn    = (bf16*)ws;  ws += (size_t)512 * 64 * 128 * 2;
    bf16* hA    = (bf16*)ws;  ws += (size_t)512 * 1024 * 2;
    bf16* hB    = (bf16*)ws;  ws += (size_t)512 * 1024 * 2;
    float* cbuf = (float*)ws; ws += (size_t)512 * 1024 * 4;
    bf16* predb = (bf16*)ws;  ws += (size_t)512 * 128 * 2;

    pack_w1<<<4096, 256, 0, stream>>>(W1, U1, Wc1t);
    pack_w2<<<4096, 256, 0, stream>>>(W2, U2, Ws2t);
    pack_wd<<<512, 256, 0, stream>>>(Wd, Wdt);
    normalize_in<<<4096, 256, 0, stream>>>(inputs, mean, var, xn);
    init_state<<<2048, 256, 0, stream>>>(hA, cbuf);

    dim3 blk(256);
    dim3 gL(8, 32), gD(8, 4);
    bf16* hin = hA; bf16* hout = hB;

    // warmup: 64 steps of cell1 over the normalized sequence
    for (int tstep = 0; tstep < 64; ++tstep) {
        lstm_step<<<gL, blk, 0, stream>>>(xn + tstep * 128, 64 * 128, 128,
                                          hin, 1024,
                                          Wc1t, 1152, 1152, b1, cbuf, hout);
        bf16* tmp = hin; hin = hout; hout = tmp;
    }
    dense_step<<<gD, blk, 0, stream>>>(hin, Wdt, bd, out, predb, 0);

    // autoregressive: cell1(pred) then cell2(h,h) with folded W2+U2
    for (int s = 1; s < 32; ++s) {
        lstm_step<<<gL, blk, 0, stream>>>(predb, 128, 128,
                                          hin, 1024,
                                          Wc1t, 1152, 1152, b1, cbuf, hout);
        { bf16* tmp = hin; hin = hout; hout = tmp; }
        lstm_step<<<gL, blk, 0, stream>>>(hin, 1024, 1024,
                                          hin, 1024,
                                          Ws2t, 1024, 1024, b2, cbuf, hout);
        { bf16* tmp = hin; hin = hout; hout = tmp; }
        dense_step<<<gD, blk, 0, stream>>>(hin, Wdt, bd, out, predb, s);
    }
}
